// SelfAttention_44074954391950
// MI455X (gfx1250) — matrix operand order repaired and checked by reading, not compiled, not run on hardware
//
#include <hip/hip_runtime.h>
#include <hip/hip_bf16.h>

// ---------------------------------------------------------------------------
// Types for WMMA / TDM operands (CDNA5 wave32)
// ---------------------------------------------------------------------------
typedef __attribute__((ext_vector_type(16))) __bf16          v16bf;
typedef __attribute__((ext_vector_type(8)))  float           v8f;
typedef __attribute__((ext_vector_type(16))) unsigned short  us16;
typedef __attribute__((ext_vector_type(8)))  unsigned short  us8;
typedef __attribute__((ext_vector_type(4)))  unsigned int    v4u;
typedef __attribute__((ext_vector_type(8)))  int             v8i;
typedef __attribute__((ext_vector_type(4)))  int             v4i;

#define B_  2
#define S_  4096
#define D_  512
#define H_  8
#define DH_ 64

__device__ __forceinline__ unsigned short f2bf(float f) {
    unsigned int u = __builtin_bit_cast(unsigned int, f);
    unsigned int r = u + 0x7FFFu + ((u >> 16) & 1u);   // round-to-nearest-even
    return (unsigned short)(r >> 16);
}

__device__ __forceinline__ us16 cat8(us8 lo, us8 hi) {
    return __builtin_shufflevector(lo, hi, 0,1,2,3,4,5,6,7,8,9,10,11,12,13,14,15);
}

__device__ __forceinline__ v8f vzero8() {
    v8f z;
#pragma unroll
    for (int i = 0; i < 8; ++i) z[i] = 0.0f;
    return z;
}

__device__ __forceinline__ v8f wmma_bf16(us16 a, us16 b, v8f c) {
    return __builtin_amdgcn_wmma_f32_16x16x32_bf16(
        false, __builtin_bit_cast(v16bf, a),
        false, __builtin_bit_cast(v16bf, b),
        (short)0, c, false, false);
}

// Load a pair-packed 16x32 bf16 A-fragment from a row-major row pointer.
__device__ __forceinline__ us16 load_afrag(const unsigned short* rowp, int kb, int half) {
    us8 lo = *(const us8*)(rowp + kb + 16 * half);
    us8 hi = *(const us8*)(rowp + kb + 16 * half + 8);
    return cat8(lo, hi);
}

// ---------------------------------------------------------------------------
// TDM: DMA a 64-row x 32-col bf16 tile of a [rows, 512] row-major matrix
// into LDS (row-major, 64x32, 4KB).  D# layout per CDNA5 ISA ch.8.
// This toolchain exposes the 6-arg builtin:
//   (v4u32 g0, v8i32 g1, v4i32 g2, v4i32 g3, v8i32 extra, i32 cpol)
// ---------------------------------------------------------------------------
__device__ __forceinline__ void
tdm_load_Btile(unsigned lds_addr, const unsigned short* gtile, unsigned rows) {
    unsigned long long ga = (unsigned long long)gtile;
    v4u g0;
    g0[0] = 1u;                                          // count=1, user mode
    g0[1] = lds_addr;                                    // LDS byte address
    g0[2] = (unsigned int)(ga & 0xFFFFFFFFull);          // global_addr lo
    g0[3] = (unsigned int)((ga >> 32) & 0x01FFFFFFull)   // global_addr hi
          | (2u << 30);                                  // type = 2 ("image")
    v8i g1;
    g1[0] = 0x00010000;                // wg_mask=0, data_size=1 (2 bytes)
    g1[1] = (int)(512u << 16);         // atomic_barrier=0, tensor_dim0=512 lo
    g1[2] = (int)(rows << 16);         // tensor_dim0 hi=0, tensor_dim1=rows lo
    g1[3] = (int)(32u << 16);          // tensor_dim1 hi=0, tile_dim0=32
    g1[4] = 64;                        // tile_dim1=64, tile_dim2=0
    g1[5] = 512;                       // tensor_dim0_stride lo (elements)
    g1[6] = 0;                         // stride hi / dim1_stride lo
    g1[7] = 0;
    v4i z4; z4[0] = 0; z4[1] = 0; z4[2] = 0; z4[3] = 0;  // 2-D tile: groups 2/3 unused
    v8i z8;
#pragma unroll
    for (int i = 0; i < 8; ++i) z8[i] = 0;
    __builtin_amdgcn_tensor_load_to_lds(g0, g1, z4, z4, z8, 0);
    __builtin_amdgcn_s_wait_tensorcnt((short)0);
}

// B fragment (16 cols x 32 k, lane-striped) from the LDS-staged 64x32 tile.
__device__ __forceinline__ us16
lds_bfrag(const unsigned short* lb, int t, int l15, int half) {
    return *(const us16*)(lb + (t * 16 + l15) * 32 + 16 * half);
}

// ---------------------------------------------------------------------------
// Kernel 0a: f32 -> bf16 elementwise
// ---------------------------------------------------------------------------
__global__ void k_conv_bf16(const float* __restrict__ in,
                            unsigned short* __restrict__ out, int n) {
    for (int i = blockIdx.x * blockDim.x + threadIdx.x; i < n;
         i += gridDim.x * blockDim.x)
        out[i] = f2bf(in[i]);
}

// Kernel 0b: f32 -> bf16 with transpose: out[c*R + r] = in[r*C + c]
__global__ void k_conv_bf16_t(const float* __restrict__ in,
                              unsigned short* __restrict__ out, int R, int C) {
    int n = R * C;
    for (int i = blockIdx.x * blockDim.x + threadIdx.x; i < n;
         i += gridDim.x * blockDim.x) {
        int r = i / C, c = i - r * C;
        out[c * R + r] = f2bf(in[i]);
    }
}

// ---------------------------------------------------------------------------
// Kernel 1: QKV GEMM.  A = x_bf16 [B*S, 512], Wt = W_in^T bf16 [1536, 512].
// All 8 waves of a block share one 64-wide N tile; the weight k-slab
// (64x32 bf16) is staged in LDS once per k-step by the Tensor Data Mover,
// then every wave computes a 32x64 tile from it (8 WMMAs / k-step / wave).
// ---------------------------------------------------------------------------
__global__ __launch_bounds__(256) void
k_qkv_gemm(const unsigned short* __restrict__ xbf,
           const unsigned short* __restrict__ Wt,
           const float* __restrict__ bin,
           unsigned short* __restrict__ Qb,
           unsigned short* __restrict__ Kb,
           unsigned short* __restrict__ Vt) {
    __shared__ unsigned short ldsB[64 * 32];             // 4KB staged W tile

    const int lane = threadIdx.x & 31;
    const int l15  = lane & 15;
    const int half = lane >> 4;
    const int wv   = threadIdx.x >> 5;
    const int nt   = blockIdx.x % 24;                    // shared by the block
    const int mg   = blockIdx.x / 24;                    // 32 m-groups
    const int mbase = (mg * 8 + wv) * 32;
    const int nbase = nt * 64;

    v8f acc[2][4];
#pragma unroll
    for (int mi = 0; mi < 2; ++mi)
#pragma unroll
        for (int t = 0; t < 4; ++t) acc[mi][t] = vzero8();

    const unsigned short* arow0 = xbf + (size_t)(mbase + l15) * 512;
    const unsigned short* arow1 = arow0 + 16 * 512;
    const unsigned lbase = (unsigned)(unsigned long long)(void*)&ldsB[0];

#pragma unroll 1
    for (int kb = 0; kb < 512; kb += 32) {
        if (wv == 0)
            tdm_load_Btile(lbase, Wt + (size_t)nbase * 512 + kb, 1536u);
        __syncthreads();                                  // tile visible

        us16 a0 = load_afrag(arow0, kb, half);
        us16 a1 = load_afrag(arow1, kb, half);
        us16 b0 = lds_bfrag(ldsB, 0, l15, half);
        us16 b1 = lds_bfrag(ldsB, 1, l15, half);
        us16 b2 = lds_bfrag(ldsB, 2, l15, half);
        us16 b3 = lds_bfrag(ldsB, 3, l15, half);

        acc[0][0] = wmma_bf16(a0, b0, acc[0][0]);
        acc[0][1] = wmma_bf16(a0, b1, acc[0][1]);
        acc[0][2] = wmma_bf16(a0, b2, acc[0][2]);
        acc[0][3] = wmma_bf16(a0, b3, acc[0][3]);
        acc[1][0] = wmma_bf16(a1, b0, acc[1][0]);
        acc[1][1] = wmma_bf16(a1, b1, acc[1][1]);
        acc[1][2] = wmma_bf16(a1, b2, acc[1][2]);
        acc[1][3] = wmma_bf16(a1, b3, acc[1][3]);
        __syncthreads();                                  // done reading tile
    }

#pragma unroll
    for (int t = 0; t < 4; ++t) {
        const int c     = nbase + t * 16 + l15;
        const float bia = bin[c];
        const int which = c >> 9;        // 0=q 1=k 2=v
        const int d     = c & 511;
        const int h     = d >> 6;
        const int dh    = d & 63;
#pragma unroll
        for (int mi = 0; mi < 2; ++mi) {
#pragma unroll
            for (int v = 0; v < 8; ++v) {
                const int grow = mbase + mi * 16 + v + 8 * half;  // row in [0, B*S)
                const int b_   = grow >> 12;
                const int s    = grow & 4095;
                const int bh   = b_ * H_ + h;
                const unsigned short bf = f2bf(acc[mi][t][v] + bia);
                if (which == 0)      Qb[((size_t)bh * S_ + s) * DH_ + dh] = bf;
                else if (which == 1) Kb[((size_t)bh * S_ + s) * DH_ + dh] = bf;
                else                 Vt[((size_t)bh * DH_ + dh) * S_ + s] = bf;
            }
        }
    }
}

// ---------------------------------------------------------------------------
// Online-softmax update for one 16-row tile over 32 keys (two 16-key frags).
// Writes exp(P) as bf16 into the per-tile LDS region Pw (16x32 row-major).
// ---------------------------------------------------------------------------
__device__ __forceinline__ void
softmax_step(v8f s0, v8f s1, int qb, int jb, int l15, int half,
             float* mrow, float* lrow, v8f* o, unsigned short* Pw) {
#pragma unroll
    for (int v = 0; v < 8; ++v) {
        const int row = qb + v + 8 * half;
        const int k0  = jb + l15;
        const int k1  = jb + 16 + l15;
        float a0 = (k0 <= row) ? s0[v] * 0.125f : -__builtin_inff();
        float a1 = (k1 <= row) ? s1[v] * 0.125f : -__builtin_inff();

        float mx = fmaxf(a0, a1);
#pragma unroll
        for (int off = 1; off < 16; off <<= 1)
            mx = fmaxf(mx, __shfl_xor(mx, off, 32));

        const float mo = mrow[v];
        const float mn = fmaxf(mo, mx);
        const float alpha = __expf(mo - mn);
        const float e0 = __expf(a0 - mn);
        const float e1 = __expf(a1 - mn);
        float ps = e0 + e1;
#pragma unroll
        for (int off = 1; off < 16; off <<= 1)
            ps += __shfl_xor(ps, off, 32);

        lrow[v] = lrow[v] * alpha + ps;
        mrow[v] = mn;
#pragma unroll
        for (int t = 0; t < 4; ++t) o[t][v] *= alpha;

        const int r32 = (v + 8 * half) * 32;
        Pw[r32 + l15]      = f2bf(e0);
        Pw[r32 + 16 + l15] = f2bf(e1);
    }
}

// ---------------------------------------------------------------------------
// Kernel 2: causal flash attention.  One wave owns TWO 16-row q-tiles of one
// (b,h): a low tile qt0 and its causal complement qt1 = 127-qt0, so every
// wave does ~the same total work and K/V fragments are shared by both tiles
// while both are active (16 WMMAs per 32-key step).
// ---------------------------------------------------------------------------
__global__ __launch_bounds__(256) void
k_flash_attn(const unsigned short* __restrict__ Qb,
             const unsigned short* __restrict__ Kb,
             const unsigned short* __restrict__ Vt,
             unsigned short* __restrict__ attnO) {
    __shared__ unsigned short Pl[8][2 * 16 * 32];   // 2KB per wave

    const int lane = threadIdx.x & 31;
    const int l15  = lane & 15;
    const int half = lane >> 4;
    const int wv   = threadIdx.x >> 5;
    // 1024 waves: 16 (b,h) x 64 tile-pairs
    const int bh  = blockIdx.x >> 3;                // 0..15
    const int qt0 = (blockIdx.x & 7) * 8 + wv;      // 0..63  (low tile)
    const int qt1 = 127 - qt0;                      // 64..127 (high tile)
    const int qb0 = qt0 << 4;
    const int qb1 = qt1 << 4;

    // Q A-fragments for both tiles
    us16 aq0[2], aq1[2];
    {
        const unsigned short* qp0 = Qb + ((size_t)bh * S_ + qb0 + l15) * DH_;
        const unsigned short* qp1 = Qb + ((size_t)bh * S_ + qb1 + l15) * DH_;
#pragma unroll
        for (int t = 0; t < 2; ++t) {
            aq0[t] = load_afrag(qp0, t * 32, half);
            aq1[t] = load_afrag(qp1, t * 32, half);
        }
    }

    float m0[8], l0[8], m1[8], l1[8];
    v8f o0[4], o1[4];
#pragma unroll
    for (int v = 0; v < 8; ++v) {
        m0[v] = -3.0e38f; l0[v] = 0.0f;
        m1[v] = -3.0e38f; l1[v] = 0.0f;
    }
#pragma unroll
    for (int t = 0; t < 4; ++t) { o0[t] = vzero8(); o1[t] = vzero8(); }

    const unsigned short* Kbh = Kb + (size_t)bh * S_ * DH_;
    const unsigned short* Vbh = Vt + (size_t)bh * DH_ * S_;
    unsigned short* Pw0 = &Pl[wv][0];
    unsigned short* Pw1 = &Pl[wv][16 * 32];

    const int nsteps = (qb1 + 16 + 31) >> 5;        // driven by the high tile
    for (int st = 0; st < nsteps; ++st) {
        const int jb = st << 5;
        const bool act0 = (jb <= qb0 + 15);         // wave-uniform

        // ---- K fragments for 32 keys (shared by both tiles) ----
        const unsigned short* kpa = Kbh + (size_t)(jb + l15) * DH_ + 16 * half;
        const unsigned short* kpb = kpa + 16 * DH_;
        us16 bk00 = *(const us16*)(kpa);
        us16 bk01 = *(const us16*)(kpa + 32);
        us16 bk10 = *(const us16*)(kpb);
        us16 bk11 = *(const us16*)(kpb + 32);

        // ---- S = Q K^T ----
        v8f s10 = vzero8(), s11 = vzero8();
        s10 = wmma_bf16(aq1[0], bk00, s10);
        s10 = wmma_bf16(aq1[1], bk01, s10);
        s11 = wmma_bf16(aq1[0], bk10, s11);
        s11 = wmma_bf16(aq1[1], bk11, s11);

        v8f s00, s01;
        if (act0) {
            s00 = vzero8(); s01 = vzero8();
            s00 = wmma_bf16(aq0[0], bk00, s00);
            s00 = wmma_bf16(aq0[1], bk01, s00);
            s01 = wmma_bf16(aq0[0], bk10, s01);
            s01 = wmma_bf16(aq0[1], bk11, s01);
        }

        // ---- online softmax + P -> LDS ----
        softmax_step(s10, s11, qb1, jb, l15, half, m1, l1, o1, Pw1);
        if (act0)
            softmax_step(s00, s01, qb0, jb, l15, half, m0, l0, o0, Pw0);

        // ---- reload P as 16x32 A-fragments ----
        us16 ap1 = cat8(*(const us8*)(Pw1 + l15 * 32 + 16 * half),
                        *(const us8*)(Pw1 + l15 * 32 + 16 * half + 8));
        us16 ap0;
        if (act0)
            ap0 = cat8(*(const us8*)(Pw0 + l15 * 32 + 16 * half),
                       *(const us8*)(Pw0 + l15 * 32 + 16 * half + 8));

        // ---- V fragments (shared) + O += P V ----
        const unsigned short* vp = Vbh + (size_t)l15 * S_ + jb + 16 * half;
        us16 bv0 = *(const us16*)(vp);
        us16 bv1 = *(const us16*)(vp + 16 * S_);
        us16 bv2 = *(const us16*)(vp + 32 * S_);
        us16 bv3 = *(const us16*)(vp + 48 * S_);

        o1[0] = wmma_bf16(ap1, bv0, o1[0]);
        o1[1] = wmma_bf16(ap1, bv1, o1[1]);
        o1[2] = wmma_bf16(ap1, bv2, o1[2]);
        o1[3] = wmma_bf16(ap1, bv3, o1[3]);
        if (act0) {
            o0[0] = wmma_bf16(ap0, bv0, o0[0]);
            o0[1] = wmma_bf16(ap0, bv1, o0[1]);
            o0[2] = wmma_bf16(ap0, bv2, o0[2]);
            o0[3] = wmma_bf16(ap0, bv3, o0[3]);
        }
    }

    // ---- epilogue: O / l -> attn buffer [B*S, 512] bf16 ----
    const int b_ = bh >> 3, h = bh & 7;
#pragma unroll
    for (int t = 0; t < 4; ++t) {
#pragma unroll
        for (int v = 0; v < 8; ++v) {
            const int r0 = qb0 + v + 8 * half;
            const int r1 = qb1 + v + 8 * half;
            attnO[((size_t)(b_ * S_ + r0)) * D_ + h * DH_ + t * 16 + l15] =
                f2bf(o0[t][v] / l0[v]);
            attnO[((size_t)(b_ * S_ + r1)) * D_ + h * DH_ + t * 16 + l15] =
                f2bf(o1[t][v] / l1[v]);
        }
    }
}

// ---------------------------------------------------------------------------
// Kernel 3: output GEMM.  out = attn_bf @ W_out + b_out, fp32 result.
// Same TDM-staged weight tile scheme as kernel 1.
// ---------------------------------------------------------------------------
__global__ __launch_bounds__(256) void
k_out_gemm(const unsigned short* __restrict__ abf,
           const unsigned short* __restrict__ Wt,
           const float* __restrict__ bout,
           float* __restrict__ out) {
    __shared__ unsigned short ldsB[64 * 32];             // 4KB staged W tile

    const int lane = threadIdx.x & 31;
    const int l15  = lane & 15;
    const int half = lane >> 4;
    const int wv   = threadIdx.x >> 5;
    const int nt   = blockIdx.x & 7;                     // shared by the block
    const int mg   = blockIdx.x >> 3;                    // 32 m-groups
    const int mbase = (mg * 8 + wv) * 32;
    const int nbase = nt * 64;

    v8f acc[2][4];
#pragma unroll
    for (int mi = 0; mi < 2; ++mi)
#pragma unroll
        for (int t = 0; t < 4; ++t) acc[mi][t] = vzero8();

    const unsigned short* arow0 = abf + (size_t)(mbase + l15) * 512;
    const unsigned short* arow1 = arow0 + 16 * 512;
    const unsigned lbase = (unsigned)(unsigned long long)(void*)&ldsB[0];

#pragma unroll 1
    for (int kb = 0; kb < 512; kb += 32) {
        if (wv == 0)
            tdm_load_Btile(lbase, Wt + (size_t)nbase * 512 + kb, 512u);
        __syncthreads();

        us16 a0 = load_afrag(arow0, kb, half);
        us16 a1 = load_afrag(arow1, kb, half);
        us16 b0 = lds_bfrag(ldsB, 0, l15, half);
        us16 b1 = lds_bfrag(ldsB, 1, l15, half);
        us16 b2 = lds_bfrag(ldsB, 2, l15, half);
        us16 b3 = lds_bfrag(ldsB, 3, l15, half);

        acc[0][0] = wmma_bf16(a0, b0, acc[0][0]);
        acc[0][1] = wmma_bf16(a0, b1, acc[0][1]);
        acc[0][2] = wmma_bf16(a0, b2, acc[0][2]);
        acc[0][3] = wmma_bf16(a0, b3, acc[0][3]);
        acc[1][0] = wmma_bf16(a1, b0, acc[1][0]);
        acc[1][1] = wmma_bf16(a1, b1, acc[1][1]);
        acc[1][2] = wmma_bf16(a1, b2, acc[1][2]);
        acc[1][3] = wmma_bf16(a1, b3, acc[1][3]);
        __syncthreads();
    }

#pragma unroll
    for (int t = 0; t < 4; ++t) {
        const int c = nbase + t * 16 + l15;
        const float bia = bout[c];
#pragma unroll
        for (int mi = 0; mi < 2; ++mi) {
#pragma unroll
            for (int v = 0; v < 8; ++v) {
                const int grow = mbase + mi * 16 + v + 8 * half;
                out[(size_t)grow * 512 + c] = acc[mi][t][v] + bia;
            }
        }
    }
}

// ---------------------------------------------------------------------------
// Host-side launch
// ---------------------------------------------------------------------------
extern "C" void kernel_launch(void* const* d_in, const int* in_sizes, int n_in,
                              void* d_out, int out_size, void* d_ws, size_t ws_size,
                              hipStream_t stream) {
    const float* x    = (const float*)d_in[0];
    const float* W_in = (const float*)d_in[1];
    const float* b_in = (const float*)d_in[2];
    const float* W_out= (const float*)d_in[3];
    const float* b_out= (const float*)d_in[4];
    float* out = (float*)d_out;

    char* ws = (char*)d_ws;
    const size_t BS = (size_t)B_ * S_;               // 8192
    const size_t SZ_XBF  = BS * D_ * 2;              // 8 MB
    const size_t SZ_WIN  = (size_t)(3 * D_) * D_ * 2;// 1.5 MB
    const size_t SZ_WOUT = (size_t)D_ * D_ * 2;      // 0.5 MB
    const size_t SZ_HEAD = (size_t)B_ * H_ * S_ * DH_ * 2; // 8 MB each

    unsigned short* xbf  = (unsigned short*)(ws);
    unsigned short* WtIn = (unsigned short*)(ws + SZ_XBF);
    unsigned short* WtOut= (unsigned short*)(ws + SZ_XBF + SZ_WIN);
    unsigned short* Qb   = (unsigned short*)(ws + SZ_XBF + SZ_WIN + SZ_WOUT);
    unsigned short* Kb   = (unsigned short*)((char*)Qb + SZ_HEAD);
    unsigned short* Vt   = (unsigned short*)((char*)Kb + SZ_HEAD);
    unsigned short* attn = (unsigned short*)((char*)Vt + SZ_HEAD);

    // Phase 0: precision conversion (+ weight transposes)
    k_conv_bf16  <<<2048, 256, 0, stream>>>(x, xbf, (int)(BS * D_));
    k_conv_bf16_t<<<768,  256, 0, stream>>>(W_in,  WtIn,  D_, 3 * D_);
    k_conv_bf16_t<<<256,  256, 0, stream>>>(W_out, WtOut, D_, D_);

    // Phase 1: QKV projection (24 N-tiles x 32 M-groups, TDM-staged weights)
    k_qkv_gemm<<<768, 256, 0, stream>>>(xbf, WtIn, b_in, Qb, Kb, Vt);

    // Phase 2: causal flash attention (1024 paired waves / 8 per block)
    k_flash_attn<<<128, 256, 0, stream>>>(Qb, Kb, Vt, attn);

    // Phase 3: output projection (8 N-tiles x 32 M-groups, TDM-staged weights)
    k_out_gemm<<<256, 256, 0, stream>>>(attn, WtOut, b_out, out);
}
